// SimpleGraphSAGELayer_83803401879831
// MI455X (gfx1250) — matrix operand order repaired and checked by reading, not compiled, not run on hardware
//
#include <hip/hip_runtime.h>
#include <hip/hip_bf16.h>
#include <stdint.h>

#define F_DIM   128
#define K_NEI   16
#define N_NODES 20000
#define KA      256          // stacked reduction dim: [x(128) | nei_mean(128)]
#define ROWS    64           // rows per workgroup
#define THREADS 256

typedef __bf16 bf16_t;
typedef __bf16 v16bf __attribute__((ext_vector_type(16)));
typedef float  v8f   __attribute__((ext_vector_type(8)));

// Split an f32 value into hi/lo bf16 (hi = RNE(v), lo = RNE(v - hi)).
static __device__ __forceinline__ void split_bf(float v, unsigned short& h, unsigned short& l) {
    bf16_t hb = (bf16_t)v;
    float  hf = (float)hb;
    bf16_t lb = (bf16_t)(v - hf);
    __builtin_memcpy(&h, &hb, 2);
    __builtin_memcpy(&l, &lb, 2);
}

// Byte offset of A element (row, f) inside the fragment-ordered LDS buffer.
// Layout: [rowtile(4)][kstep(8)][lane(32)][32 bytes], where for lane = 16*khalf + m:
//   bytes  0..31 hold K = 32*ks + 16*khalf + 0..15   (VGPR0-7)
static __device__ __forceinline__ int frag_off(int row, int f) {
    int rt = row >> 4, m = row & 15;
    int ks = f >> 5, rem = f & 31;
    int half_sel = (rem >> 4) & 1;
    int sub = rem & 15;
    int khalf = half_sel;
    int elem = sub;
    int lane = khalf * 16 + m;
    return (((rt * 8 + ks) * 32 + lane) << 5) + (elem << 1);
}

// ---------------------------------------------------------------------------
// Prep: build transposed split-bf16 stacked weights Wt[n][k] (n<128, k<256)
// with Wt[n][k] = W_self[k][n] (k<128) else W_nei[k-128][n]; and bsum = b_self+b_nei.
// B-fragment for WMMA: lane (khalf,n) reads 16 contiguous bf16 at k = 32*ks + 16*khalf.
// ---------------------------------------------------------------------------
__global__ __launch_bounds__(256) void sage_prep(
    const float* __restrict__ W_self, const float* __restrict__ b_self,
    const float* __restrict__ W_nei,  const float* __restrict__ b_nei,
    unsigned short* __restrict__ wt_hi, unsigned short* __restrict__ wt_lo,
    float* __restrict__ bsum)
{
    const int idx = blockIdx.x * 256 + threadIdx.x;   // 0..32767
    const int k = idx >> 7;
    const int n = idx & 127;

    unsigned short h, l;
    split_bf(W_self[idx], h, l);            // W_self[k*128+n]
    wt_hi[n * KA + k] = h;
    wt_lo[n * KA + k] = l;

    split_bf(W_nei[idx], h, l);
    wt_hi[n * KA + 128 + k] = h;
    wt_lo[n * KA + 128 + k] = l;

    if (blockIdx.x == 0 && threadIdx.x < F_DIM)
        bsum[threadIdx.x] = b_self[threadIdx.x] + b_nei[threadIdx.x];
}

// ---------------------------------------------------------------------------
// Main: gather + mean into LDS (WMMA fragment order, split bf16), then
// D = [x | nei] x [W_self ; W_nei] via v_wmma_f32_16x16x32_bf16 (hi/lo split),
// bias + relu epilogue.
// ---------------------------------------------------------------------------
__global__ __launch_bounds__(THREADS) void sage_main(
    const float* __restrict__ x, const long long* __restrict__ adj,
    const unsigned short* __restrict__ wt_hi_u, const unsigned short* __restrict__ wt_lo_u,
    const float* __restrict__ bsum, float* __restrict__ out)
{
    __shared__ __attribute__((aligned(32))) unsigned char sAhi[32768];
    __shared__ __attribute__((aligned(32))) unsigned char sAlo[32768];

    const int tid  = threadIdx.x;
    const int base = blockIdx.x * ROWS;

    // ---------------- Phase 1: gather + mean, write split-bf16 fragments ----
    {
        const int r  = tid >> 2;               // row within block, 0..63
        const int fo = (tid & 3) * 32;         // feature chunk, 0/32/64/96
        const int gr = base + r;               // global row
        const int b  = gr / N_NODES;           // batch index
        const long long* arow = adj + (long long)gr * K_NEI;
        const float* xrow = x + (long long)gr * F_DIM + fo;
        const float* xbat = x + (long long)b * N_NODES * F_DIM + fo;

        float xv[32];
        #pragma unroll
        for (int i = 0; i < 8; ++i) {
            float4 t = ((const float4*)xrow)[i];
            xv[4*i+0] = t.x; xv[4*i+1] = t.y; xv[4*i+2] = t.z; xv[4*i+3] = t.w;
        }

        float acc[32];
        #pragma unroll
        for (int i = 0; i < 32; ++i) acc[i] = 0.0f;

        for (int j = 0; j < K_NEI; ++j) {
            const int nb = (int)arow[j];
            const float4* nrow = (const float4*)(xbat + (long long)nb * F_DIM);
            #pragma unroll
            for (int i = 0; i < 8; ++i) {
                float4 t = nrow[i];
                acc[4*i+0] += t.x; acc[4*i+1] += t.y;
                acc[4*i+2] += t.z; acc[4*i+3] += t.w;
            }
        }

        const float inv_k = 1.0f / (float)K_NEI;
        #pragma unroll
        for (int i = 0; i < 32; i += 2) {
            unsigned short h0, l0, h1, l1;
            // x region: A columns [0,128)
            split_bf(xv[i],   h0, l0);
            split_bf(xv[i+1], h1, l1);
            int off = frag_off(r, fo + i);
            *(unsigned int*)(sAhi + off) = (unsigned)h0 | ((unsigned)h1 << 16);
            *(unsigned int*)(sAlo + off) = (unsigned)l0 | ((unsigned)l1 << 16);
            // nei region: A columns [128,256)
            split_bf(acc[i]   * inv_k, h0, l0);
            split_bf(acc[i+1] * inv_k, h1, l1);
            off = frag_off(r, 128 + fo + i);
            *(unsigned int*)(sAhi + off) = (unsigned)h0 | ((unsigned)h1 << 16);
            *(unsigned int*)(sAlo + off) = (unsigned)l0 | ((unsigned)l1 << 16);
        }
    }
    __syncthreads();

    // ---------------- Phase 2: WMMA GEMM -----------------------------------
    const int lane  = tid & 31;
    const int w     = tid >> 5;     // wave 0..7
    const int rt    = w >> 1;       // row tile 0..3 (16 rows each)
    const int ch    = w & 1;        // column half (64 cols each)
    const int khalf = lane >> 4;
    const int nlo   = lane & 15;

    const bf16_t* wt_hi = (const bf16_t*)wt_hi_u;
    const bf16_t* wt_lo = (const bf16_t*)wt_lo_u;

    v8f c0 = {0.f,0.f,0.f,0.f,0.f,0.f,0.f,0.f};
    v8f acc[4] = {c0, c0, c0, c0};

    #pragma unroll
    for (int ks = 0; ks < 8; ++ks) {
        const int aoff = ((rt * 8 + ks) * 32 + lane) << 5;
        const v16bf a_hi = *(const v16bf*)(sAhi + aoff);
        const v16bf a_lo = *(const v16bf*)(sAlo + aoff);
        #pragma unroll
        for (int ct = 0; ct < 4; ++ct) {
            const int n    = ch * 64 + ct * 16 + nlo;
            const int woff = n * KA + ks * 32 + khalf * 16;
            const v16bf b_hi = *(const v16bf*)(wt_hi + woff);
            const v16bf b_lo = *(const v16bf*)(wt_lo + woff);
            acc[ct] = __builtin_amdgcn_wmma_f32_16x16x32_bf16(
                false, a_hi, false, b_hi, (short)0, acc[ct], false, false);
            acc[ct] = __builtin_amdgcn_wmma_f32_16x16x32_bf16(
                false, a_hi, false, b_lo, (short)0, acc[ct], false, false);
            acc[ct] = __builtin_amdgcn_wmma_f32_16x16x32_bf16(
                false, a_lo, false, b_hi, (short)0, acc[ct], false, false);
        }
    }

    // ---------------- Epilogue: bias + relu + store ------------------------
    #pragma unroll
    for (int ct = 0; ct < 4; ++ct) {
        const int n = ch * 64 + ct * 16 + nlo;
        const float bias = bsum[n];
        #pragma unroll
        for (int rr = 0; rr < 8; ++rr) {
            const int gr = base + rt * 16 + rr + 8 * khalf;  // C/D: M = vgpr + 8*khalf
            float v = acc[ct][rr] + bias;
            out[(long long)gr * F_DIM + n] = v > 0.0f ? v : 0.0f;
        }
    }
}

// ---------------------------------------------------------------------------
extern "C" void kernel_launch(void* const* d_in, const int* in_sizes, int n_in,
                              void* d_out, int out_size, void* d_ws, size_t ws_size,
                              hipStream_t stream) {
    const float*     x      = (const float*)d_in[0];
    const long long* adj    = (const long long*)d_in[1];
    const float*     W_self = (const float*)d_in[2];
    const float*     b_self = (const float*)d_in[3];
    const float*     W_nei  = (const float*)d_in[4];
    const float*     b_nei  = (const float*)d_in[5];
    float* out = (float*)d_out;

    // workspace layout: Wt_hi (64KB) | Wt_lo (64KB) | bsum (512B)
    unsigned short* wt_hi = (unsigned short*)d_ws;
    unsigned short* wt_lo = (unsigned short*)((char*)d_ws + 65536);
    float*          bsum  = (float*)((char*)d_ws + 131072);

    sage_prep<<<128, 256, 0, stream>>>(W_self, b_self, W_nei, b_nei, wt_hi, wt_lo, bsum);

    const int total_rows = in_sizes[0] / F_DIM;   // B*N = 80000
    sage_main<<<total_rows / ROWS, THREADS, 0, stream>>>(x, adj, wt_hi, wt_lo, bsum, out);
}
